// LocalAttention_10282151707569
// MI455X (gfx1250) — matrix operand order repaired and checked by reading, not compiled, not run on hardware
//
#include <hip/hip_runtime.h>
#include <hip/hip_bf16.h>

typedef _Float16 v16h __attribute__((ext_vector_type(16)));
typedef _Float16 v8h  __attribute__((ext_vector_type(8)));
typedef float    v8f  __attribute__((ext_vector_type(8)));

#define EMB    1024
#define HEADS  16
#define HD     64
#define NSEQ   2048
#define BATCH  2
#define NT     (NSEQ / 16)        // 128 key tiles per sequence
#define MROWS  (BATCH * NSEQ)     // 4096
#define WIN    128
#define SCALE  0.03125f           // 1/sqrt(1024)

static __device__ __forceinline__ v8f wmma16(v16h a, v16h b, v8f c) {
    return __builtin_amdgcn_wmma_f32_16x16x32_f16(
        false, a, false, b, (short)0, c, false, false);
}

// A-fragment (16x32 f16, MxK): lane holds row = lane&15; elements 0..7 are
// K = g*8..g*8+7, elements 8..15 are K = 16+g*8.., g = lane>>4. K-contiguous.
static __device__ __forceinline__ v16h load_a_frag(const _Float16* base, int ld,
                                                   int row, int k0, int lane) {
    int g = lane >> 4;
    const _Float16* p = base + (size_t)row * ld + k0 + g * 8;
    v8h lo = *(const v8h*)p;
    v8h hi = *(const v8h*)(p + 16);
    v16h a;
#pragma unroll
    for (int i = 0; i < 8; ++i) { a[i] = lo[i]; a[8 + i] = hi[i]; }
    return a;
}

// B-fragment (32x16 f16, KxN): lane holds col n = lane&15; elements 0..7 are
// K = g*8.., elements 8..15 are K = 16+g*8.. (row `n` of source).
static __device__ __forceinline__ v16h load_b_frag(const _Float16* base, int ld,
                                                   int n_row, int k0, int lane) {
    int g = lane >> 4;
    const _Float16* p = base + (size_t)n_row * ld + k0 + g * 8;
    v8h lo = *(const v8h*)p;
    v8h hi = *(const v8h*)(p + 16);
    v16h b;
#pragma unroll
    for (int i = 0; i < 8; ++i) { b[i] = lo[i]; b[8 + i] = hi[i]; }
    return b;
}

static __device__ __forceinline__ float redmax16(float x) {
    x = fmaxf(x, __shfl_xor(x, 1, 32));
    x = fmaxf(x, __shfl_xor(x, 2, 32));
    x = fmaxf(x, __shfl_xor(x, 4, 32));
    x = fmaxf(x, __shfl_xor(x, 8, 32));
    return x;
}
static __device__ __forceinline__ float redsum16(float x) {
    x += __shfl_xor(x, 1, 32);
    x += __shfl_xor(x, 2, 32);
    x += __shfl_xor(x, 4, 32);
    x += __shfl_xor(x, 8, 32);
    return x;
}

// LDS byte-offset of a generic pointer known to point into shared memory.
static __device__ __forceinline__ unsigned ldsoff(const void* p) {
    const __attribute__((address_space(3))) char* p3 =
        (const __attribute__((address_space(3))) char*)p;
    return (unsigned)(unsigned long long)p3;
}

// gfx1250 async global->LDS copy of 16 bytes per lane (ASYNCcnt-tracked).
static __device__ __forceinline__ void async_cp16(unsigned lds_off,
                                                  const _Float16* gp) {
    unsigned long long ga = (unsigned long long)gp;
    asm volatile("global_load_async_to_lds_b128 %0, %1, off"
                 :: "v"(lds_off), "v"(ga)
                 : "memory");
}

// ---------------- f32 -> f16 conversion ----------------
__global__ void __launch_bounds__(256) cvt_f32_f16(const float* __restrict__ s,
                                                   _Float16* __restrict__ d, int n) {
    int i = blockIdx.x * 256 + threadIdx.x;
    if (i < n) d[i] = (_Float16)s[i];
}

// ---------------- LDS-staged projection GEMM: y = A @ W^T + bias ----------------
// A: [4096 x 1024] f16 (K-contig), W: [1024 x 1024] f16 (K-contig rows).
// Block = 8 waves computes a 64x128 tile; A/B panels double-buffered in LDS,
// filled with global_load_async_to_lds_b128, consumed via ds_load_b128 frags.
// mode 0: out f16 [B,H,N,d]   (Q, K)
// mode 1: out f16 [B,H,d,N]   (V transposed)
// mode 2: out f32 [M, F]      (final projection -> d_out)
__global__ void __launch_bounds__(256) gemm_proj(const _Float16* __restrict__ A,
                                                 const _Float16* __restrict__ W,
                                                 const float* __restrict__ bias,
                                                 void* __restrict__ out, int mode) {
    __shared__ __align__(16) _Float16 sA[2][64 * 64];    // 16 KB
    __shared__ __align__(16) _Float16 sB[2][128 * 64];   // 32 KB

    int tid  = threadIdx.x;
    int lane = tid & 31;
    int wv   = tid >> 5;
    int bm = blockIdx.x >> 3;            // 64 M-blocks of 64 rows
    int bf = blockIdx.x & 7;             // 8  F-blocks of 128 cols
    int m0b = bm * 64, f0b = bf * 128;
    int ra = (wv >> 2) * 32;             // wave's local A-row base (0/32)
    int rb = (wv & 3) * 32;              // wave's local B-row base (0..96)
    int m0 = m0b + ra, f0 = f0b + rb;
    int g = lane >> 4, cl = lane & 15;

    // stage one 64-wide K-panel of A (64x64) and B (128x64) into LDS buffer
    auto stage = [&](int buf, int k0) {
        unsigned baseA = ldsoff(&sA[buf][0]);
#pragma unroll
        for (int i = 0; i < 2; ++i) {
            int c = tid + 256 * i;              // 512 chunks of 16B
            int row = c >> 3, cw = c & 7;
            async_cp16(baseA + (unsigned)(row * 64 + cw * 8) * 2,
                       A + (size_t)(m0b + row) * EMB + k0 + cw * 8);
        }
        unsigned baseB = ldsoff(&sB[buf][0]);
#pragma unroll
        for (int i = 0; i < 4; ++i) {
            int c = tid + 256 * i;              // 1024 chunks of 16B
            int row = c >> 3, cw = c & 7;
            async_cp16(baseB + (unsigned)(row * 64 + cw * 8) * 2,
                       W + (size_t)(f0b + row) * EMB + k0 + cw * 8);
        }
    };

    v8f c00 = {}, c01 = {}, c10 = {}, c11 = {};

    stage(0, 0);
    for (int t = 0; t < EMB / 64; ++t) {
        asm volatile("s_wait_asynccnt 0" ::: "memory");
        __syncthreads();
        if (t + 1 < EMB / 64) stage((t + 1) & 1, (t + 1) * 64);

        const _Float16* tA = &sA[t & 1][0];
        const _Float16* tB = &sB[t & 1][0];
#pragma unroll
        for (int kk = 0; kk < 64; kk += 32) {
            v16h a0 = load_a_frag(tA, 64, ra + cl, kk, lane);
            v16h a1 = load_a_frag(tA, 64, ra + 16 + cl, kk, lane);
            v16h b0 = load_b_frag(tB, 64, rb + cl, kk, lane);
            v16h b1 = load_b_frag(tB, 64, rb + 16 + cl, kk, lane);
            c00 = wmma16(a0, b0, c00);
            c01 = wmma16(a0, b1, c01);
            c10 = wmma16(a1, b0, c10);
            c11 = wmma16(a1, b1, c11);
        }
        __syncthreads();
    }

    float bv0 = bias[f0 + cl];
    float bv1 = bias[f0 + 16 + cl];
    _Float16* outh = (_Float16*)out;
    float*    outf = (float*)out;

#pragma unroll
    for (int ar = 0; ar < 2; ++ar) {
#pragma unroll
        for (int cg = 0; cg < 2; ++cg) {
            v8f cc = ar == 0 ? (cg == 0 ? c00 : c01) : (cg == 0 ? c10 : c11);
            float bv = cg == 0 ? bv0 : bv1;
            int f = f0 + cg * 16 + cl;
#pragma unroll
            for (int v = 0; v < 8; ++v) {
                int m = m0 + ar * 16 + g * 8 + v;
                float val = cc[v] + bv;
                if (mode == 2) {
                    outf[(size_t)m * EMB + f] = val;
                } else {
                    int b = m >> 11, n = m & (NSEQ - 1);
                    int h = f >> 6, dv = f & (HD - 1);
                    if (mode == 0)
                        outh[(((size_t)(b * HEADS + h)) * NSEQ + n) * HD + dv] = (_Float16)val;
                    else
                        outh[(((size_t)(b * HEADS + h)) * HD + dv) * NSEQ + n] = (_Float16)val;
                }
            }
        }
    }
}

// ---------------- windowed flash attention ----------------
// One wave per (b, h, 16-query tile). Keys iterated in pairs of 16-tiles
// (32 keys = one f16 WMMA K-dim). Online softmax in f32.
__global__ void __launch_bounds__(256) attn_window(const _Float16* __restrict__ Qh,
                                                   const _Float16* __restrict__ Kh,
                                                   const _Float16* __restrict__ Vt,
                                                   _Float16* __restrict__ Oh) {
    __shared__ __align__(16) _Float16 lds[8][16 * 32];

    int lane = threadIdx.x & 31;
    int wv   = threadIdx.x >> 5;
    int wid  = blockIdx.x * 8 + wv;              // [0, 4096)
    int b  = wid >> 11;
    int h  = (wid >> 7) & (HEADS - 1);
    int t0 = wid & (NT - 1);
    int q0 = t0 << 4;
    int g = lane >> 4, cl = lane & 15;

    const _Float16* Qb = Qh + (((size_t)(b * HEADS + h)) * NSEQ + q0) * HD;
    const _Float16* Kb = Kh + ((size_t)(b * HEADS + h)) * NSEQ * HD;
    const _Float16* Vb = Vt + ((size_t)(b * HEADS + h)) * HD * NSEQ;

    v16h qa0 = load_a_frag(Qb, HD, cl, 0, lane);
    v16h qa1 = load_a_frag(Qb, HD, cl, 32, lane);

    v8f oc0 = {}, oc1 = {}, oc2 = {}, oc3 = {};
    float mrow[8], lrow[8];
#pragma unroll
    for (int v = 0; v < 8; ++v) { mrow[v] = -3.0e38f; lrow[v] = 0.f; }

    int ktS = t0 - 8 < 0 ? 0 : t0 - 8;
    int ktE = t0 + 8 > NT - 1 ? NT - 1 : t0 + 8;
    int npairs = (ktE - ktS + 2) >> 1;

    for (int p = 0; p < npairs; ++p) {
        int kt0 = ktS + 2 * p;
        int kt1 = kt0 + 1;
        bool have1 = kt1 < NT;

        v16h kb00 = load_b_frag(Kb, HD, kt0 * 16 + cl, 0, lane);
        v16h kb01 = load_b_frag(Kb, HD, kt0 * 16 + cl, 32, lane);
        v16h zf = {};
        v16h kb10 = have1 ? load_b_frag(Kb, HD, kt1 * 16 + cl, 0, lane) : zf;
        v16h kb11 = have1 ? load_b_frag(Kb, HD, kt1 * 16 + cl, 32, lane) : zf;

        v8f s0 = {}, s1 = {};
        s0 = wmma16(qa0, kb00, s0);
        s0 = wmma16(qa1, kb01, s0);
        s1 = wmma16(qa0, kb10, s1);
        s1 = wmma16(qa1, kb11, s1);

        float alpha[8];
#pragma unroll
        for (int v = 0; v < 8; ++v) {
            int r  = g * 8 + v;
            int qi = q0 + r;
            int j0 = (kt0 << 4) + cl;
            int j1 = (kt1 << 4) + cl;
            int d0 = qi - j0; if (d0 < 0) d0 = -d0;
            int d1 = qi - j1; if (d1 < 0) d1 = -d1;
            float x0 = (d0 <= WIN) ? s0[v] * SCALE : -3.0e38f;
            float x1 = (have1 && d1 <= WIN) ? s1[v] * SCALE : -3.0e38f;
            float mx = redmax16(fmaxf(x0, x1));
            float mn = fmaxf(mrow[v], mx);
            float a  = __expf(mrow[v] - mn);
            float p0 = __expf(x0 - mn);
            float p1 = __expf(x1 - mn);
            float sm = redsum16(p0 + p1);
            lrow[v] = lrow[v] * a + sm;
            mrow[v] = mn;
            alpha[v] = a;
            lds[wv][r * 32 + cl]      = (_Float16)p0;
            lds[wv][r * 32 + 16 + cl] = (_Float16)p1;
        }
#pragma unroll
        for (int v = 0; v < 8; ++v) {
            oc0[v] *= alpha[v]; oc1[v] *= alpha[v];
            oc2[v] *= alpha[v]; oc3[v] *= alpha[v];
        }

        asm volatile("s_wait_dscnt 0" ::: "memory");

        const _Float16* lp = &lds[wv][cl * 32 + g * 8];
        v8h plo = *(const v8h*)lp;
        v8h phi = *(const v8h*)(lp + 16);
        v16h pa;
#pragma unroll
        for (int i = 0; i < 8; ++i) { pa[i] = plo[i]; pa[8 + i] = phi[i]; }

#pragma unroll
        for (int cc = 0; cc < 4; ++cc) {
            const _Float16* vp = Vb + (size_t)(cc * 16 + cl) * NSEQ + (kt0 << 4) + g * 8;
            v8h lo, hi;
            lo = *(const v8h*)vp;
            if (!have1) {
#pragma unroll
                for (int i = 0; i < 8; ++i) { hi[i] = (_Float16)0; }
            } else {
                hi = *(const v8h*)(vp + 16);
            }
            v16h vb;
#pragma unroll
            for (int i = 0; i < 8; ++i) { vb[i] = lo[i]; vb[8 + i] = hi[i]; }
            if (cc == 0) oc0 = wmma16(pa, vb, oc0);
            else if (cc == 1) oc1 = wmma16(pa, vb, oc1);
            else if (cc == 2) oc2 = wmma16(pa, vb, oc2);
            else oc3 = wmma16(pa, vb, oc3);
        }
    }

#pragma unroll
    for (int v = 0; v < 8; ++v) {
        int r = g * 8 + v;
        float inv = 1.0f / lrow[v];
        size_t rowbase = ((size_t)b * NSEQ + q0 + r) * EMB + h * HD;
        Oh[rowbase + 0 * 16 + cl]  = (_Float16)(oc0[v] * inv);
        Oh[rowbase + 1 * 16 + cl]  = (_Float16)(oc1[v] * inv);
        Oh[rowbase + 2 * 16 + cl]  = (_Float16)(oc2[v] * inv);
        Oh[rowbase + 3 * 16 + cl]  = (_Float16)(oc3[v] * inv);
    }
}

extern "C" void kernel_launch(void* const* d_in, const int* in_sizes, int n_in,
                              void* d_out, int out_size, void* d_ws, size_t ws_size,
                              hipStream_t stream) {
    const float* x  = (const float*)d_in[0];
    const float* Wq = (const float*)d_in[1];
    const float* bq = (const float*)d_in[2];
    const float* Wk = (const float*)d_in[3];
    const float* bk = (const float*)d_in[4];
    const float* Wv = (const float*)d_in[5];
    const float* bv = (const float*)d_in[6];
    const float* Wp = (const float*)d_in[7];
    const float* bp = (const float*)d_in[8];

    const size_t NX = (size_t)MROWS * EMB;   // 4M elements
    const size_t NW = (size_t)EMB * EMB;     // 1M elements

    char* w = (char*)d_ws;
    size_t off = 0;
    auto carve = [&](size_t elems) {
        _Float16* p = (_Float16*)(w + off);
        off = (off + elems * sizeof(_Float16) + 255) & ~(size_t)255;
        return p;
    };
    _Float16* xh  = carve(NX);
    _Float16* Wqh = carve(NW);
    _Float16* Wkh = carve(NW);
    _Float16* Wvh = carve(NW);
    _Float16* Wph = carve(NW);
    _Float16* Qh  = carve(NX);
    _Float16* Kh  = carve(NX);
    _Float16* Vt  = carve(NX);
    _Float16* Oh  = carve(NX);
    (void)ws_size;

    // 1) convert inputs to f16
    cvt_f32_f16<<<(int)((NX + 255) / 256), 256, 0, stream>>>(x,  xh,  (int)NX);
    cvt_f32_f16<<<(int)((NW + 255) / 256), 256, 0, stream>>>(Wq, Wqh, (int)NW);
    cvt_f32_f16<<<(int)((NW + 255) / 256), 256, 0, stream>>>(Wk, Wkh, (int)NW);
    cvt_f32_f16<<<(int)((NW + 255) / 256), 256, 0, stream>>>(Wv, Wvh, (int)NW);
    cvt_f32_f16<<<(int)((NW + 255) / 256), 256, 0, stream>>>(Wp, Wph, (int)NW);

    // 2) Q/K/V projections (LDS-staged WMMA GEMM), 512 blocks x 8 waves
    gemm_proj<<<512, 256, 0, stream>>>(xh, Wqh, bq, (void*)Qh, 0);
    gemm_proj<<<512, 256, 0, stream>>>(xh, Wkh, bk, (void*)Kh, 0);
    gemm_proj<<<512, 256, 0, stream>>>(xh, Wvh, bv, (void*)Vt, 1);

    // 3) windowed flash attention, one wave per (b, h, query-tile)
    attn_window<<<512, 256, 0, stream>>>(Qh, Kh, Vt, Oh);

    // 4) output projection -> f32 d_out
    gemm_proj<<<512, 256, 0, stream>>>(Oh, Wph, bp, d_out, 2);
}
